// Seq2seqLuongAttention_55327768707599
// MI455X (gfx1250) — matrix operand order repaired and checked by reading, not compiled, not run on hardware
//
#include <hip/hip_runtime.h>
#include <hip/hip_bf16.h>
#include <math.h>

// ---------------------------------------------------------------------------
// Types for CDNA5 WMMA (wave32): A/B are 16 bf16 per lane, C/D are 8 f32.
// ---------------------------------------------------------------------------
typedef __attribute__((ext_vector_type(16))) __bf16          v16bf;
typedef __attribute__((ext_vector_type(8)))  float           v8f;
typedef __attribute__((ext_vector_type(16))) unsigned short  v16us;
typedef __attribute__((ext_vector_type(8)))  unsigned short  v8us;

#define DEVFN static __device__ __forceinline__

DEVFN unsigned short f2bf(float f) {
    unsigned int u = __float_as_uint(f);
    u += 0x7FFFu + ((u >> 16) & 1u);          // round-to-nearest-even
    return (unsigned short)(u >> 16);
}
DEVFN float bf2f(unsigned short h) { return __uint_as_float(((unsigned int)h) << 16); }

// A-fragment (16x32 bf16, row-major source, lda in elements).
// Layout: lane L row = L%16; half = L/16;
//   elements 0..15 <- K = kbase + half*16 + 0..15
DEVFN v16bf load_a_frag(const unsigned short* A, int lda, int row0, int kbase, int lane) {
    int row  = row0 + (lane & 15);
    int half = lane >> 4;
    const unsigned short* p = A + (size_t)row * lda + kbase + half * 16;
    v8us lo = *(const v8us*)p;
    v8us hi = *(const v8us*)(p + 8);
    v16us r;
#pragma unroll
    for (int i = 0; i < 8; ++i) { r[i] = lo[i]; r[i + 8] = hi[i]; }
    return __builtin_bit_cast(v16bf, r);
}

// B-fragment from pre-packed tile: 32 lanes * 16 contiguous bf16 (1 KB/tile).
DEVFN v16bf load_b_tile(const unsigned short* tile, int lane) {
    v16us r = *(const v16us*)(tile + lane * 16);
    return __builtin_bit_cast(v16bf, r);
}

DEVFN v8f wmma_bf16(v16bf a, v16bf b, v8f c) {
    return __builtin_amdgcn_wmma_f32_16x16x32_bf16(false, a, false, b, (short)0, c, false, false);
}

DEVFN float sigmoidf_(float x) { return 1.0f / (1.0f + __expf(-x)); }

// ---------------------------------------------------------------------------
// Pack a row-major fp32 weight (K x N) into bf16 B-fragment tiles.
// Tile index t = ntile*ktiles + kt; lane L stores 16 bf16 contiguously:
//   col n = ntile*16 + L%16; K = kt*32 + (L/16)*16 + i, i=0..15
// ---------------------------------------------------------------------------
__global__ void pack_b_kernel(const float* __restrict__ W, unsigned short* __restrict__ out,
                              int K, int N) {
    int wave = (int)((blockIdx.x * blockDim.x + threadIdx.x) >> 5);
    int lane = threadIdx.x & 31;
    int ktiles = K >> 5, ntiles = N >> 4;
    if (wave >= ktiles * ntiles) return;
    int ntile = wave / ktiles, kt = wave % ktiles;
    int n     = ntile * 16 + (lane & 15);
    int kbase = kt * 32 + (lane >> 4) * 16;
    unsigned short* dst = out + (size_t)wave * 512 + lane * 16;
#pragma unroll
    for (int i = 0; i < 16; ++i)
        dst[i] = f2bf(W[(size_t)(kbase + i) * N + n]);
}

// ---------------------------------------------------------------------------
// Embedding gather -> per-step row-major bf16 activations out[(s*B+b)*E + e]
// ---------------------------------------------------------------------------
__global__ void gather_x_kernel(const int* __restrict__ seq, const float* __restrict__ emb,
                                unsigned short* __restrict__ out, int B, int S, int E) {
    size_t idx = (size_t)blockIdx.x * blockDim.x + threadIdx.x;
    size_t total = (size_t)B * S * E;
    if (idx >= total) return;
    int e = (int)(idx % E);
    size_t t = idx / E;
    int b = (int)(t % B);
    int s = (int)(t / B);
    int tok = seq[b * S + s];
    out[idx] = f2bf(emb[(size_t)tok * E + e]);
}

// ---------------------------------------------------------------------------
// Fused GRU step: gates = [xA0|xA1] @ Wx + b + h @ Wh, then nonlinearity.
// 128 wave-tasks: jtile (0..63) x mtile (0..1). 4 WMMA accumulators per wave.
// All three k-loops are register double-buffered: iteration kt issues the
// loads for kt+1 BEFORE consuming kt's fragments, so each v_wmma only has to
// wait for the older load group (partial s_wait_loadcnt instead of 0).
// extra_mode: 1 -> enc_hiddens[b][step][h], 2 -> decout row (b*64+step).
// ---------------------------------------------------------------------------
__global__ void gru_step_kernel(
    const unsigned short* __restrict__ xA0, int ldx0, int kx0tiles,
    const unsigned short* __restrict__ xA1, int ldx1, int kx1tiles,
    const unsigned short* __restrict__ hA,
    const unsigned short* __restrict__ Wx_p,
    const unsigned short* __restrict__ Wh_p,
    const float* __restrict__ bias,
    const float* __restrict__ h_in,
    float* __restrict__ h_out,
    unsigned short* __restrict__ h_out_bf,
    unsigned short* __restrict__ extra, int extra_mode, int step) {
    const int H = 1024;
    int wave = (int)((blockIdx.x * blockDim.x + threadIdx.x) >> 5);
    int lane = threadIdx.x & 31;
    if (wave >= 128) return;
    int mtile = wave & 1;
    int jtile = wave >> 1;
    int wx_ktiles = kx0tiles + kx1tiles;

    v8f accZ = {}, accR = {}, accXH = {}, accHH = {};

    const unsigned short* WzBase = Wx_p + (size_t)(jtile)       * wx_ktiles * 512;
    const unsigned short* WrBase = Wx_p + (size_t)(64  + jtile) * wx_ktiles * 512;
    const unsigned short* WcBase = Wx_p + (size_t)(128 + jtile) * wx_ktiles * 512;

    // ---- x part 0 @ Wx (k-tiles 0 .. kx0tiles-1), pipelined ----
    {
        v16bf a0  = load_a_frag(xA0, ldx0, mtile * 16, 0, lane);
        v16bf bz0 = load_b_tile(WzBase, lane);
        v16bf br0 = load_b_tile(WrBase, lane);
        v16bf bc0 = load_b_tile(WcBase, lane);
        for (int kt = 0; kt < kx0tiles; ++kt) {
            int ktn = (kt + 1 < kx0tiles) ? kt + 1 : kt;   // uniform; branchless
            v16bf a1  = load_a_frag(xA0, ldx0, mtile * 16, ktn * 32, lane);
            v16bf bz1 = load_b_tile(WzBase + (size_t)ktn * 512, lane);
            v16bf br1 = load_b_tile(WrBase + (size_t)ktn * 512, lane);
            v16bf bc1 = load_b_tile(WcBase + (size_t)ktn * 512, lane);
            accZ  = wmma_bf16(a0, bz0, accZ);
            accR  = wmma_bf16(a0, br0, accR);
            accXH = wmma_bf16(a0, bc0, accXH);
            a0 = a1; bz0 = bz1; br0 = br1; bc0 = bc1;
        }
    }
    // ---- x part 1 (ctx) @ Wx (k-tiles kx0tiles .. wx_ktiles-1), pipelined ----
    if (kx1tiles > 0) {
        v16bf a0  = load_a_frag(xA1, ldx1, mtile * 16, 0, lane);
        v16bf bz0 = load_b_tile(WzBase + (size_t)kx0tiles * 512, lane);
        v16bf br0 = load_b_tile(WrBase + (size_t)kx0tiles * 512, lane);
        v16bf bc0 = load_b_tile(WcBase + (size_t)kx0tiles * 512, lane);
        for (int kt = 0; kt < kx1tiles; ++kt) {
            int ktn = (kt + 1 < kx1tiles) ? kt + 1 : kt;
            int ktg = kx0tiles + ktn;
            v16bf a1  = load_a_frag(xA1, ldx1, mtile * 16, ktn * 32, lane);
            v16bf bz1 = load_b_tile(WzBase + (size_t)ktg * 512, lane);
            v16bf br1 = load_b_tile(WrBase + (size_t)ktg * 512, lane);
            v16bf bc1 = load_b_tile(WcBase + (size_t)ktg * 512, lane);
            accZ  = wmma_bf16(a0, bz0, accZ);
            accR  = wmma_bf16(a0, br0, accR);
            accXH = wmma_bf16(a0, bc0, accXH);
            a0 = a1; bz0 = bz1; br0 = br1; bc0 = bc1;
        }
    }
    // ---- h @ Wh (32 k-tiles); candidate part kept separate (accHH) ----
    {
        const unsigned short* HzBase = Wh_p + (size_t)(jtile)       * 32 * 512;
        const unsigned short* HrBase = Wh_p + (size_t)(64  + jtile) * 32 * 512;
        const unsigned short* HcBase = Wh_p + (size_t)(128 + jtile) * 32 * 512;
        v16bf a0  = load_a_frag(hA, H, mtile * 16, 0, lane);
        v16bf bz0 = load_b_tile(HzBase, lane);
        v16bf br0 = load_b_tile(HrBase, lane);
        v16bf bc0 = load_b_tile(HcBase, lane);
        for (int kt = 0; kt < 32; ++kt) {
            int ktn = (kt + 1 < 32) ? kt + 1 : kt;
            v16bf a1  = load_a_frag(hA, H, mtile * 16, ktn * 32, lane);
            v16bf bz1 = load_b_tile(HzBase + (size_t)ktn * 512, lane);
            v16bf br1 = load_b_tile(HrBase + (size_t)ktn * 512, lane);
            v16bf bc1 = load_b_tile(HcBase + (size_t)ktn * 512, lane);
            accZ  = wmma_bf16(a0, bz0, accZ);
            accR  = wmma_bf16(a0, br0, accR);
            accHH = wmma_bf16(a0, bc0, accHH);
            a0 = a1; bz0 = bz1; br0 = br1; bc0 = bc1;
        }
    }
    // Epilogue: C layout -> lane col = lane%16, row = vgpr + 8*(lane/16)
    int colL = lane & 15, half = lane >> 4;
    int j = jtile * 16 + colL;
    float bz = bias[j], brb = bias[H + j], bcb = bias[2 * H + j];
#pragma unroll
    for (int r = 0; r < 8; ++r) {
        int m = mtile * 16 + r + 8 * half;
        float z    = sigmoidf_(accZ[r] + bz);
        float rr   = sigmoidf_(accR[r] + brb);
        float cand = tanhf(accXH[r] + bcb + rr * accHH[r]);
        float hold = h_in[(size_t)m * H + j];
        float hnew = z * hold + (1.0f - z) * cand;
        h_out[(size_t)m * H + j] = hnew;
        unsigned short hb = f2bf(hnew);
        h_out_bf[(size_t)m * H + j] = hb;
        if (extra_mode == 1)
            extra[((size_t)m * 256 + step) * H + j] = hb;           // enc_hiddens[b][s][h]
        else if (extra_mode == 2)
            extra[((size_t)(m * 64 + step)) * H + j] = hb;          // decout row b*64+s
    }
}

// ---------------------------------------------------------------------------
// Generic small GEMM: C(bf16, M x N) = A(bf16 row-major) @ B(packed tiles)
// One wave per output tile, register double-buffered k-loop.
// Used for s = h @ W_att (2x64 tiles).
// ---------------------------------------------------------------------------
__global__ void gemm_bf16_kernel(const unsigned short* __restrict__ A, int lda,
                                 const unsigned short* __restrict__ Bp, int ktiles,
                                 unsigned short* __restrict__ C, int ldc,
                                 int Mtiles, int Ntiles) {
    int wave = (int)((blockIdx.x * blockDim.x + threadIdx.x) >> 5);
    int lane = threadIdx.x & 31;
    if (wave >= Mtiles * Ntiles) return;
    int mtile = wave % Mtiles;
    int ntile = wave / Mtiles;
    v8f acc = {};
    const unsigned short* Bbase = Bp + (size_t)ntile * ktiles * 512;
    v16bf a0 = load_a_frag(A, lda, mtile * 16, 0, lane);
    v16bf b0 = load_b_tile(Bbase, lane);
    for (int kt = 0; kt < ktiles; ++kt) {
        int ktn = (kt + 1 < ktiles) ? kt + 1 : kt;
        v16bf a1 = load_a_frag(A, lda, mtile * 16, ktn * 32, lane);
        v16bf b1 = load_b_tile(Bbase + (size_t)ktn * 512, lane);
        acc = wmma_bf16(a0, b0, acc);
        a0 = a1; b0 = b1;
    }
    int colL = lane & 15, half = lane >> 4;
#pragma unroll
    for (int r = 0; r < 8; ++r) {
        int row = mtile * 16 + r + 8 * half;
        C[(size_t)row * ldc + ntile * 16 + colL] = f2bf(acc[r]);
    }
}

// ---------------------------------------------------------------------------
// Attention scores: scores[b][s] = sum_h enc[b][s][h] * s_vec[b][h]
// One wave per (b, s) dot product of length 1024 (VALU; FLOPs are tiny).
// ---------------------------------------------------------------------------
__global__ void attn_scores_kernel(const unsigned short* __restrict__ enc,
                                   const unsigned short* __restrict__ svec,
                                   float* __restrict__ scores) {
    int wave = (int)((blockIdx.x * blockDim.x + threadIdx.x) >> 5);
    int lane = threadIdx.x & 31;
    if (wave >= 32 * 256) return;
    int b = wave >> 8, si = wave & 255;
    const unsigned short* e = enc + ((size_t)b * 256 + si) * 1024;
    const unsigned short* s = svec + (size_t)b * 1024;
    float acc = 0.0f;
#pragma unroll
    for (int it = 0; it < 4; ++it) {
        int k = it * 256 + lane * 8;
        v8us ev = *(const v8us*)(e + k);
        v8us sv = *(const v8us*)(s + k);
#pragma unroll
        for (int i = 0; i < 8; ++i) acc += bf2f(ev[i]) * bf2f(sv[i]);
    }
#pragma unroll
    for (int off = 16; off > 0; off >>= 1) acc += __shfl_down(acc, off, 32);
    if (lane == 0) scores[b * 256 + si] = acc;
}

// Softmax over S=256 per batch row; one 256-thread block per b.
__global__ void softmax_kernel(const float* __restrict__ scores, float* __restrict__ attn) {
    __shared__ float red[256];
    int b = blockIdx.x, t = threadIdx.x;
    float v = scores[b * 256 + t];
    red[t] = v;
    __syncthreads();
    for (int o = 128; o > 0; o >>= 1) { if (t < o) red[t] = fmaxf(red[t], red[t + o]); __syncthreads(); }
    float mx = red[0];
    __syncthreads();
    float e = __expf(v - mx);
    red[t] = e;
    __syncthreads();
    for (int o = 128; o > 0; o >>= 1) { if (t < o) red[t] += red[t + o]; __syncthreads(); }
    attn[b * 256 + t] = e / red[0];
}

// ctx[b][h] = sum_s attn[b][s] * enc[b][s][h]; coalesced across h.
__global__ void attn_ctx_kernel(const float* __restrict__ attn,
                                const unsigned short* __restrict__ enc,
                                unsigned short* __restrict__ ctx_bf) {
    int idx = (int)(blockIdx.x * blockDim.x + threadIdx.x);
    if (idx >= 32 * 1024) return;
    int b = idx >> 10, h = idx & 1023;
    const unsigned short* e = enc + (size_t)b * 256 * 1024 + h;
    const float* a = attn + b * 256;
    float acc = 0.0f;
    for (int s = 0; s < 256; ++s) acc += a[s] * bf2f(e[(size_t)s * 1024]);
    ctx_bf[(size_t)b * 1024 + h] = f2bf(acc);
}

// ---------------------------------------------------------------------------
// Output projection: logits(2048 x 32000, f32) = decout(2048x1024 bf16) @
// W_out(packed) + b_out. One wave per 16x16 tile; 128 x 2000 tiles.
// k-loop register double-buffered; one global_prefetch of the next n-tile's
// weight block per wave warms L2 for the 65 MB weight stream. The kernel is
// otherwise bound by the 262 MB logits store (~11 us at 23.3 TB/s).
// ---------------------------------------------------------------------------
__global__ void gemm_out_kernel(const unsigned short* __restrict__ A,
                                const unsigned short* __restrict__ Bp,
                                const float* __restrict__ bias,
                                float* __restrict__ C) {
    int wave = (int)((blockIdx.x * blockDim.x + threadIdx.x) >> 5);
    int lane = threadIdx.x & 31;
    if (wave >= 128 * 2000) return;
    int ntile = wave % 2000;
    int mtile = wave / 2000;
    v8f acc = {};
    const unsigned short* Bbase = Bp + (size_t)ntile * 32 * 512;
    // Warm L2 with the next n-tile's packed weights (CDNA5 global_prefetch).
    if (ntile + 1 < 2000)
        __builtin_prefetch(Bbase + (size_t)32 * 512 + lane * 16, 0, 1);
    v16bf a0 = load_a_frag(A, 1024, mtile * 16, 0, lane);
    v16bf b0 = load_b_tile(Bbase, lane);
    for (int kt = 0; kt < 32; ++kt) {
        int ktn = (kt + 1 < 32) ? kt + 1 : kt;
        v16bf a1 = load_a_frag(A, 1024, mtile * 16, ktn * 32, lane);
        v16bf b1 = load_b_tile(Bbase + (size_t)ktn * 512, lane);
        acc = wmma_bf16(a0, b0, acc);
        a0 = a1; b0 = b1;
    }
    int colL = lane & 15, half = lane >> 4;
    int col = ntile * 16 + colL;
    float bv = bias[col];
#pragma unroll
    for (int r = 0; r < 8; ++r) {
        int row = mtile * 16 + r + 8 * half;
        C[(size_t)row * 32000 + col] = acc[r] + bv;
    }
}

// ---------------------------------------------------------------------------
// Host launch
// ---------------------------------------------------------------------------
extern "C" void kernel_launch(void* const* d_in, const int* in_sizes, int n_in,
                              void* d_out, int out_size, void* d_ws, size_t ws_size,
                              hipStream_t stream) {
    (void)in_sizes; (void)n_in; (void)out_size; (void)ws_size;
    const int B = 32, S = 256, SD = 64, E = 512, H = 1024;

    const int*   src_seq = (const int*)d_in[0];
    const int*   dec_seq = (const int*)d_in[1];
    const float* emb     = (const float*)d_in[2];
    const float* Wx_enc  = (const float*)d_in[3];
    const float* Wh_enc  = (const float*)d_in[4];
    const float* b_enc   = (const float*)d_in[5];
    const float* Wx_dec  = (const float*)d_in[6];
    const float* Wh_dec  = (const float*)d_in[7];
    const float* b_dec   = (const float*)d_in[8];
    const float* W_att   = (const float*)d_in[9];
    const float* W_out   = (const float*)d_in[10];
    const float* b_out   = (const float*)d_in[11];
    float* logits = (float*)d_out;

    // Workspace carving (256B aligned).
    char* base = (char*)d_ws;
    size_t off = 0;
    auto carve = [&](size_t bytes) -> char* {
        char* p = base + off;
        off = (off + bytes + 255) & ~(size_t)255;
        return p;
    };
    unsigned short* wxe_p  = (unsigned short*)carve((size_t)16 * 192  * 512 * 2);
    unsigned short* whe_p  = (unsigned short*)carve((size_t)32 * 192  * 512 * 2);
    unsigned short* wxd_p  = (unsigned short*)carve((size_t)48 * 192  * 512 * 2);
    unsigned short* whd_p  = (unsigned short*)carve((size_t)32 * 192  * 512 * 2);
    unsigned short* watt_p = (unsigned short*)carve((size_t)32 * 64   * 512 * 2);
    unsigned short* wout_p = (unsigned short*)carve((size_t)32 * 2000 * 512 * 2);
    unsigned short* srcx   = (unsigned short*)carve((size_t)S  * B * E * 2);
    unsigned short* decx   = (unsigned short*)carve((size_t)SD * B * E * 2);
    float*          hf[2]  = { (float*)carve((size_t)B * H * 4), (float*)carve((size_t)B * H * 4) };
    unsigned short* hb[2]  = { (unsigned short*)carve((size_t)B * H * 2),
                               (unsigned short*)carve((size_t)B * H * 2) };
    unsigned short* enc_bf = (unsigned short*)carve((size_t)B * S * H * 2);
    unsigned short* s_bf   = (unsigned short*)carve((size_t)B * H * 2);
    float*          scores = (float*)carve((size_t)B * S * 4);
    float*          attn   = (float*)carve((size_t)B * S * 4);
    unsigned short* ctx_bf = (unsigned short*)carve((size_t)B * H * 2);
    unsigned short* decout = (unsigned short*)carve((size_t)B * SD * H * 2);

    // --- 1. Pack all weights into bf16 WMMA tiles (once per call) ---
    auto pack = [&](const float* W, unsigned short* out, int K, int N) {
        int waves = (K >> 5) * (N >> 4);
        int blocks = (waves + 7) / 8;
        hipLaunchKernelGGL(pack_b_kernel, dim3(blocks), dim3(256), 0, stream, W, out, K, N);
    };
    pack(Wx_enc, wxe_p, 512, 3072);
    pack(Wh_enc, whe_p, 1024, 3072);
    pack(Wx_dec, wxd_p, 1536, 3072);
    pack(Wh_dec, whd_p, 1024, 3072);
    pack(W_att, watt_p, 1024, 1024);
    pack(W_out, wout_p, 1024, 32000);

    // --- 2. Embedding gathers -> per-step bf16 activation matrices ---
    {
        size_t n1 = (size_t)B * S * E;
        hipLaunchKernelGGL(gather_x_kernel, dim3((unsigned)((n1 + 255) / 256)), dim3(256), 0,
                           stream, src_seq, emb, srcx, B, S, E);
        size_t n2 = (size_t)B * SD * E;
        hipLaunchKernelGGL(gather_x_kernel, dim3((unsigned)((n2 + 255) / 256)), dim3(256), 0,
                           stream, dec_seq, emb, decx, B, SD, E);
    }

    // --- 3. Encoder: 256 sequential fused GRU steps ---
    int cur = 0;
    hipMemsetAsync(hf[0], 0, (size_t)B * H * 4, stream);
    hipMemsetAsync(hb[0], 0, (size_t)B * H * 2, stream);
    for (int s = 0; s < S; ++s) {
        hipLaunchKernelGGL(gru_step_kernel, dim3(16), dim3(256), 0, stream,
                           srcx + (size_t)s * B * E, E, 16,
                           (const unsigned short*)nullptr, 0, 0,
                           hb[cur], wxe_p, whe_p, b_enc,
                           hf[cur], hf[1 - cur], hb[1 - cur],
                           enc_bf, 1, s);
        cur ^= 1;
    }

    // --- 4. Decoder: attention + fused GRU per step ---
    for (int t = 0; t < SD; ++t) {
        // s = h @ W_att  (32x1024 @ 1024x1024), bf16 out
        hipLaunchKernelGGL(gemm_bf16_kernel, dim3(16), dim3(256), 0, stream,
                           hb[cur], H, watt_p, 32, s_bf, H, 2, 64);
        hipLaunchKernelGGL(attn_scores_kernel, dim3(1024), dim3(256), 0, stream,
                           enc_bf, s_bf, scores);
        hipLaunchKernelGGL(softmax_kernel, dim3(32), dim3(256), 0, stream, scores, attn);
        hipLaunchKernelGGL(attn_ctx_kernel, dim3(128), dim3(256), 0, stream,
                           attn, enc_bf, ctx_bf);
        // GRU on concat([x_t, ctx]) : k-tiles 0..15 from x, 16..47 from ctx
        hipLaunchKernelGGL(gru_step_kernel, dim3(16), dim3(256), 0, stream,
                           decx + (size_t)t * B * E, E, 16,
                           ctx_bf, H, 32,
                           hb[cur], wxd_p, whd_p, b_dec,
                           hf[cur], hf[1 - cur], hb[1 - cur],
                           decout, 2, t);
        cur ^= 1;
    }

    // --- 5. Output projection: 2048x1024 @ 1024x32000 + b_out -> fp32 logits ---
    hipLaunchKernelGGL(gemm_out_kernel, dim3(32000), dim3(256), 0, stream,
                       decout, wout_p, b_out, logits);
}
